// UnifiedAutoencoderNet_65481071401694
// MI455X (gfx1250) — matrix operand order repaired and checked by reading, not compiled, not run on hardware
//
#include <hip/hip_runtime.h>

typedef __attribute__((ext_vector_type(8)))  __bf16 v8bf;
typedef __attribute__((ext_vector_type(16))) __bf16 v16bf;
typedef __attribute__((ext_vector_type(8)))  float  v8f;

#define BM 128
#define BN 128
#define BK 64
#define LDT 72   // padded LDS row stride (bf16 elems, 144B): conflict-free frag reads

__device__ __forceinline__ unsigned int f2bf_u(float f) {
  unsigned int u = __float_as_uint(f);
  return (u + 0x7FFFu + ((u >> 16) & 1u)) >> 16;  // RNE bf16
}
__device__ __forceinline__ float bf2f(unsigned short h) {
  return __uint_as_float(((unsigned int)h) << 16);
}

// CDNA5 async global->LDS copy, 16B per lane, tracked by ASYNCcnt.
// lds = wave-relative LDS byte address (low 32 bits of the generic pointer),
// g   = 64-bit global address.
__device__ __forceinline__ void async_b128(unsigned lds, const void* g) {
  asm volatile("global_load_async_to_lds_b128 %0, %1, off"
               :: "v"(lds), "v"((unsigned long long)(size_t)g)
               : "memory");
}
#define S_WAIT_ASYNCCNT(immstr) asm volatile("s_wait_asynccnt " immstr ::: "memory")

// A fragment (16x32 MxK bf16): lane l, row=l&15; halves 0..7 -> K=kh8.., 8..15 -> K=kh8+16..
__device__ __forceinline__ v16bf frag_a(const unsigned short* p) {
  v8bf lo = *(const v8bf*)(p);
  v8bf hi = *(const v8bf*)(p + 16);
  v16bf r;
#pragma unroll
  for (int i = 0; i < 8; ++i) { r[i] = lo[i]; r[i + 8] = hi[i]; }
  return r;
}
// B fragment (32x16 KxN bf16): lane l, col=l&15; halves 0..7 -> K=kh8.., 8..15 -> K=kh8+16..
__device__ __forceinline__ v16bf frag_b(const unsigned short* p) {
  v8bf lo = *(const v8bf*)(p);
  v8bf hi = *(const v8bf*)(p + 16);
  v16bf r;
#pragma unroll
  for (int i = 0; i < 8; ++i) { r[i] = lo[i]; r[i + 8] = hi[i]; }
  return r;
}

// out[m][n] = act( sum_k X[m][k]*W[n][k] + bias[n] ); X,W bf16 row-major [.,K].
// Output bf16 (OUTF32=0) or fp32 (OUTF32=1). M,N multiples of 128, K multiple of 64.
template<int N, int K, int RELU, int OUTF32>
__global__ __launch_bounds__(256) void gemm_bf16(
    const unsigned short* __restrict__ X,
    const unsigned short* __restrict__ W,
    const float* __restrict__ bias,
    void* __restrict__ outp)
{
  static_assert(K % BK == 0, "K must be multiple of 64");
  constexpr int ITERS = K / BK;
  __shared__ unsigned short sA[2][BM * LDT];
  __shared__ unsigned short sB[2][BN * LDT];

  const int tid    = threadIdx.x;
  const int wave   = tid >> 5;
  const int lane   = tid & 31;
  const int blockM = blockIdx.y * BM;
  const int blockN = blockIdx.x * BN;
  const int waveM  = (wave & 3) * 32;
  const int waveN  = (wave >> 2) * 64;
  const int r15    = lane & 15;
  const int kh8    = (lane >> 4) * 8;
  const int kh16   = (lane >> 4) * 16;

  // 128x64 bf16 tile = 1024 16B chunks; 4 chunks per thread per matrix.
  unsigned offT[4];
  const unsigned short* xp[4];
  const unsigned short* wp[4];
#pragma unroll
  for (int i = 0; i < 4; ++i) {
    int idx = tid + i * 256;
    int r   = idx >> 3;         // tile row
    int c   = (idx & 7) * 8;    // k offset (elements)
    offT[i] = (unsigned)((r * LDT + c) * 2);   // LDS byte offset within a stage
    xp[i] = X + (size_t)(blockM + r) * K + c;
    wp[i] = W + (size_t)(blockN + r) * K + c;
  }
  unsigned baseA[2] = { (unsigned)(size_t)&sA[0][0], (unsigned)(size_t)&sA[1][0] };
  unsigned baseB[2] = { (unsigned)(size_t)&sB[0][0], (unsigned)(size_t)&sB[1][0] };

  const v8f vz = {};
  v8f acc[2][4];
#pragma unroll
  for (int mi = 0; mi < 2; ++mi)
#pragma unroll
    for (int ni = 0; ni < 4; ++ni)
      acc[mi][ni] = vz;

  // prologue: fill stage 0 asynchronously (no VGPR data staging)
#pragma unroll
  for (int i = 0; i < 4; ++i) async_b128(baseA[0] + offT[i], xp[i]);
#pragma unroll
  for (int i = 0; i < 4; ++i) async_b128(baseB[0] + offT[i], wp[i]);

  for (int it = 0; it < ITERS; ++it) {
    const int s = it & 1;
    if (it + 1 < ITERS) {
      // issue next stage's fills; they overlap the WMMA burst below
#pragma unroll
      for (int i = 0; i < 4; ++i) {
        xp[i] += BK; wp[i] += BK;
        async_b128(baseA[s ^ 1] + offT[i], xp[i]);
        async_b128(baseB[s ^ 1] + offT[i], wp[i]);
      }
      // ASYNCcnt completes in order: <=8 outstanding => current stage (first 8) done
      S_WAIT_ASYNCCNT("0x8");
    } else {
      S_WAIT_ASYNCCNT("0x0");
    }
    __syncthreads();   // all waves' fills for stage s visible

    // two k-steps of 32 -> 16 WMMAs per wave per iteration
#pragma unroll
    for (int ks = 0; ks < 2; ++ks) {
      const int ko = ks * 32;
      v16bf af[2], bfr[4];
#pragma unroll
      for (int mi = 0; mi < 2; ++mi)
        af[mi] = frag_a(&sA[s][(waveM + mi * 16 + r15) * LDT + ko + kh8]);
#pragma unroll
      for (int ni = 0; ni < 4; ++ni)
        bfr[ni] = frag_b(&sB[s][(waveN + ni * 16 + r15) * LDT + ko + kh8]);
#pragma unroll
      for (int mi = 0; mi < 2; ++mi)
#pragma unroll
        for (int ni = 0; ni < 4; ++ni)
          acc[mi][ni] = __builtin_amdgcn_wmma_f32_16x16x32_bf16(
              false, af[mi], false, bfr[ni], (short)0, acc[mi][ni], false, false);
    }
    __syncthreads();   // all reads of stage s done before it is refilled
  }

  // epilogue: bias + optional ReLU; C/D layout: VGPR v, lane l -> row=v+8*(l>>4), col=l&15
  const int hi8 = (lane >> 4) * 8;
#pragma unroll
  for (int ni = 0; ni < 4; ++ni) {
    int col  = blockN + waveN + ni * 16 + r15;
    float bv = bias[col];
#pragma unroll
    for (int mi = 0; mi < 2; ++mi) {
      int row0 = blockM + waveM + mi * 16 + hi8;
#pragma unroll
      for (int v = 0; v < 8; ++v) {
        float val = acc[mi][ni][v] + bv;
        if (RELU) val = fmaxf(val, 0.0f);
        if (OUTF32)
          ((float*)outp)[(size_t)(row0 + v) * N + col] = val;
        else
          ((unsigned short*)outp)[(size_t)(row0 + v) * N + col] =
              (unsigned short)f2bf_u(val);
      }
    }
  }
}

// fp32 -> bf16 bulk conversion; n multiple of 8
__global__ __launch_bounds__(256) void cvt_f32_bf16(
    const float* __restrict__ src, unsigned short* __restrict__ dst, long n)
{
  long i = ((long)blockIdx.x * 256 + threadIdx.x) * 8;
  if (i >= n) return;
  float4 a = *(const float4*)(src + i);
  float4 b = *(const float4*)(src + i + 4);
  uint4 p;
  p.x = f2bf_u(a.x) | (f2bf_u(a.y) << 16);
  p.y = f2bf_u(a.z) | (f2bf_u(a.w) << 16);
  p.z = f2bf_u(b.x) | (f2bf_u(b.y) << 16);
  p.w = f2bf_u(b.z) | (f2bf_u(b.w) << 16);
  *(uint4*)(dst + i) = p;
}

// Wd1 (1024x16 fp32) -> zero-padded (1024x64 bf16)
__global__ __launch_bounds__(256) void cvt_pad_wd1(
    const float* __restrict__ src, unsigned short* __restrict__ dst)
{
  int idx = blockIdx.x * 256 + threadIdx.x;   // 1024*64
  int r = idx >> 6, c = idx & 63;
  float v = (c < 16) ? src[r * 16 + c] : 0.0f;
  dst[idx] = (unsigned short)f2bf_u(v);
}

// Per-qubit circuit on bf16 latent (B,256) -> refined (B,64) bf16, cols 16..63 = 0
__global__ __launch_bounds__(256) void quantum_kernel(
    const unsigned short* __restrict__ latent, const float* __restrict__ P,
    unsigned short* __restrict__ refined)
{
  int idx = blockIdx.x * 256 + threadIdx.x;   // 8192*64
  int b = idx >> 6, q = idx & 63;
  if (q >= 16) { refined[idx] = 0; return; }
  float th = bf2f(latent[(size_t)b * 256 + q]);
  float ar = __cosf(th * 0.5f), ai = 0.f;
  float br = __sinf(th * 0.5f), bi = 0.f;
#pragma unroll
  for (int l = 0; l < 2; ++l) {
    float t = P[(l * 16 + q) * 3 + 0] * 0.5f;       // RX
    float c = __cosf(t), s = __sinf(t);
    float nar = c * ar + s * bi, nai = c * ai - s * br;
    float nbr = c * br + s * ai, nbi = c * bi - s * ar;
    ar = nar; ai = nai; br = nbr; bi = nbi;
    t = P[(l * 16 + q) * 3 + 1] * 0.5f;             // RY
    c = __cosf(t); s = __sinf(t);
    nar = c * ar - s * br; nai = c * ai - s * bi;
    nbr = s * ar + c * br; nbi = s * ai + c * bi;
    ar = nar; ai = nai; br = nbr; bi = nbi;
    t = P[(l * 16 + q) * 3 + 2] * 0.5f;             // RZ
    c = __cosf(t); s = __sinf(t);
    nar = c * ar + s * ai; nai = c * ai - s * ar;
    nbr = c * br - s * bi; nbi = c * bi + s * br;
    ar = nar; ai = nai; br = nbr; bi = nbi;
  }
  float z = (ar * ar + ai * ai) - (br * br + bi * bi);
  refined[idx] = (unsigned short)f2bf_u(z);
}

extern "C" void kernel_launch(void* const* d_in, const int* in_sizes, int n_in,
                              void* d_out, int out_size, void* d_ws, size_t ws_size,
                              hipStream_t stream) {
  (void)in_sizes; (void)n_in; (void)out_size; (void)ws_size;
  const float* x   = (const float*)d_in[0];
  const float* We1 = (const float*)d_in[1];
  const float* be1 = (const float*)d_in[2];
  const float* We2 = (const float*)d_in[3];
  const float* be2 = (const float*)d_in[4];
  const float* We3 = (const float*)d_in[5];
  const float* be3 = (const float*)d_in[6];
  const float* P   = (const float*)d_in[7];
  const float* Wd1 = (const float*)d_in[8];
  const float* bd1 = (const float*)d_in[9];
  const float* Wd2 = (const float*)d_in[10];
  const float* bd2 = (const float*)d_in[11];
  const float* Wd3 = (const float*)d_in[12];
  const float* bd3 = (const float*)d_in[13];
  float* out = (float*)d_out;

  char* ws = (char*)d_ws;
  const size_t MB = 1024 * 1024;
  // region 0..64MB: xb (dead after L1) then reused for h2 / latent / refined / d1
  unsigned short* xb      = (unsigned short*)(ws);              // 8192*4096 bf16 (64MB)
  unsigned short* h2      = (unsigned short*)(ws);              // 8192*1024 bf16 (16MB)
  unsigned short* latent  = (unsigned short*)(ws + 16 * MB);    // 8192*256  bf16 ( 4MB)
  unsigned short* refined = (unsigned short*)(ws + 20 * MB);    // 8192*64   bf16 ( 1MB)
  unsigned short* d1      = (unsigned short*)(ws + 24 * MB);    // 8192*1024 bf16 (16MB)
  // weights (persistent): 64..106MB
  unsigned short* wb1  = (unsigned short*)(ws + 64 * MB);       // 2048*4096 (16MB)
  unsigned short* wb2  = (unsigned short*)(ws + 80 * MB);       // 1024*2048 ( 4MB)
  unsigned short* wb3  = (unsigned short*)(ws + 84 * MB);       // 256*1024  (0.5MB)
  unsigned short* wd1p = (unsigned short*)(ws + 85 * MB);       // 1024*64   (128KB)
  unsigned short* wd2b = (unsigned short*)(ws + 86 * MB);       // 2048*1024 ( 4MB)
  unsigned short* wd3b = (unsigned short*)(ws + 90 * MB);       // 4096*2048 (16MB)
  // region 106..138MB: h1 (dead after L2) then reused for d2
  unsigned short* h1 = (unsigned short*)(ws + 106 * MB);        // 8192*2048 (32MB)
  unsigned short* d2 = (unsigned short*)(ws + 106 * MB);        // 8192*2048 (32MB)

  dim3 blk(256);
  auto cvt = [&](const float* s, unsigned short* d, long n) {
    cvt_f32_bf16<<<dim3((unsigned)((n / 8 + 255) / 256)), blk, 0, stream>>>(s, d, n);
  };
  // one-time precision conversion (memory-bound pre-pass)
  cvt(x,   xb,   (long)8192 * 4096);
  cvt(We1, wb1,  (long)2048 * 4096);
  cvt(We2, wb2,  (long)1024 * 2048);
  cvt(We3, wb3,  (long)256  * 1024);
  cvt(Wd2, wd2b, (long)2048 * 1024);
  cvt(Wd3, wd3b, (long)4096 * 2048);
  cvt_pad_wd1<<<dim3((1024 * 64) / 256), blk, 0, stream>>>(Wd1, wd1p);

  gemm_bf16<2048, 4096, 1, 0><<<dim3(2048 / BN, 8192 / BM), blk, 0, stream>>>(xb, wb1, be1, h1);
  gemm_bf16<1024, 2048, 1, 0><<<dim3(1024 / BN, 8192 / BM), blk, 0, stream>>>(h1, wb2, be2, h2);
  gemm_bf16< 256, 1024, 0, 0><<<dim3( 256 / BN, 8192 / BM), blk, 0, stream>>>(h2, wb3, be3, latent);
  quantum_kernel<<<dim3((8192 * 64) / 256), blk, 0, stream>>>(latent, P, refined);
  gemm_bf16<1024,   64, 1, 0><<<dim3(1024 / BN, 8192 / BM), blk, 0, stream>>>(refined, wd1p, bd1, d1);
  gemm_bf16<2048, 1024, 1, 0><<<dim3(2048 / BN, 8192 / BM), blk, 0, stream>>>(d1, wd2b, bd2, d2);
  gemm_bf16<4096, 2048, 0, 1><<<dim3(4096 / BN, 8192 / BM), blk, 0, stream>>>(d2, wd3b, bd3, out);
}
